// EdgeCNN_Net_70970039599211
// MI455X (gfx1250) — matrix operand order repaired and checked by reading, not compiled, not run on hardware
//
#include <hip/hip_runtime.h>
#include <cstdint>
#include <cstddef>

// ---------------------------------------------------------------------------
// CDNA5 (gfx1250, wave32) EdgeConv: bf16 WMMA 16x16x32, f32 accumulate.
// ---------------------------------------------------------------------------

typedef __attribute__((ext_vector_type(16))) __bf16 v16bf;
typedef __attribute__((ext_vector_type(8)))  float  v8f;

union Frag { v16bf v; uint4 q[2]; };

#define FEAT_ROWB 528                  // 256 bf16 (512B) + 16B pad
#define H1_ROWB   272                  // 128 bf16 (256B) + 16B pad
#define WAVE_LDS  (16 * FEAT_ROWB + 16 * H1_ROWB)   // 12800 B per wave
#define WAVES_PER_BLOCK 4

__device__ __forceinline__ unsigned short f2bf(float f) {
  unsigned int u = __float_as_uint(f);
  u += 0x7FFFu + ((u >> 16) & 1u);     // round-to-nearest-even
  return (unsigned short)(u >> 16);
}

// Float atomic max via signed-int max / unsigned-int min (works for any sign mix;
// lowers to native global_atomic_max_i32 / global_atomic_min_u32).
__device__ __forceinline__ void atomicMaxF(float* addr, float v) {
  if (v >= 0.0f) atomicMax((int*)addr, __float_as_int(v));
  else           atomicMin((unsigned int*)addr, __float_as_uint(v));
}

// ---------------------------------------------------------------------------
// Pre-swizzle a KxN (N=128) fp32 weight matrix into bf16 WMMA B-fragment order.
// Fragment f = kc*8+nt holds 32 lanes x 16 bf16: lane L -> column n = nt*16+(L&15),
// K-values kc*32 + (t>>3)*16 + (L>>4)*8 + (t&7) (matches A table-order layout).
// ---------------------------------------------------------------------------
__global__ void swizzleW_bf16(const float* __restrict__ W,
                              unsigned short* __restrict__ out, int K) {
  int idx = blockIdx.x * blockDim.x + threadIdx.x;
  int total = K * 128;
  if (idx >= total) return;
  int t  = idx & 15;
  int L  = (idx >> 4) & 31;
  int f  = idx >> 9;
  int nt = f & 7;
  int kc = f >> 3;
  int n  = nt * 16 + (L & 15);
  int k  = kc * 32 + ((t >> 3) << 4) + ((L >> 4) << 3) + (t & 7);
  out[idx] = f2bf(W[k * 128 + n]);
}

__global__ void fill_u32(unsigned int* __restrict__ p, unsigned int v, size_t n) {
  size_t i = (size_t)blockIdx.x * blockDim.x + threadIdx.x;
  if (i < n) p[i] = v;
}

__global__ void fixup_neginf(float* __restrict__ p, int n) {
  int i = blockIdx.x * blockDim.x + threadIdx.x;
  if (i < n && __float_as_uint(p[i]) == 0xFF800000u) p[i] = 0.0f;
}

// ---------------------------------------------------------------------------
// Fused EdgeConv layer (H=128): per 16-edge tile per wave:
//   feat = [x_i | x_j - x_i]  (16 x 256 bf16 in LDS)
//   h1   = relu(feat @ W1 + b1)   -- 8x8 WMMA, bf16 back to LDS
//   msg  = h1 @ W2 + b2           -- 4x8 WMMA
//   out[dst] = atomic_max(out[dst], msg)   (out pre-filled with 0 => fused relu)
// ---------------------------------------------------------------------------
__global__ __launch_bounds__(128) void edgeconv_layer_kernel(
    const float* __restrict__ xin,
    const int* __restrict__ srcp, const int* __restrict__ dstp,
    const unsigned short* __restrict__ W1sw, const float* __restrict__ b1,
    const unsigned short* __restrict__ W2sw, const float* __restrict__ b2,
    float* __restrict__ out, int nE)
{
  __shared__ uint4 smem_raw[(WAVES_PER_BLOCK * WAVE_LDS) / 16];
  const int lane = threadIdx.x & 31;
  const int wv   = threadIdx.x >> 5;
  unsigned char* base  = (unsigned char*)smem_raw + wv * WAVE_LDS;
  unsigned char* featB = base;
  unsigned char* h1B   = base + 16 * FEAT_ROWB;

  const int tile   = blockIdx.x * WAVES_PER_BLOCK + wv;
  const int e0     = tile * 16;
  const bool active = (e0 < nE);        // wave-uniform

  // ---- stage feat tile: each lane covers 4 consecutive features per edge ----
  if (active) {
    for (int e = 0; e < 16; ++e) {
      int ge = e0 + e; if (ge >= nE) ge = nE - 1;
      const int gi = dstp[ge];
      const int gj = srcp[ge];
      const float4 xi = ((const float4*)(xin + (size_t)gi * 128))[lane];
      const float4 xj = ((const float4*)(xin + (size_t)gj * 128))[lane];
      uint2 a, d;
      a.x = (unsigned int)f2bf(xi.x)        | ((unsigned int)f2bf(xi.y)        << 16);
      a.y = (unsigned int)f2bf(xi.z)        | ((unsigned int)f2bf(xi.w)        << 16);
      d.x = (unsigned int)f2bf(xj.x - xi.x) | ((unsigned int)f2bf(xj.y - xi.y) << 16);
      d.y = (unsigned int)f2bf(xj.z - xi.z) | ((unsigned int)f2bf(xj.w - xi.w) << 16);
      unsigned char* row = featB + e * FEAT_ROWB;
      *(uint2*)(row + lane * 8)       = a;   // x_i       -> features [0,128)
      *(uint2*)(row + 256 + lane * 8) = d;   // x_j - x_i -> features [128,256)
    }
  }
  __syncthreads();

  v8f acc[8];
  const int m   = lane & 15;           // A-matrix row for this lane
  const int off = (lane >> 4) << 3;    // K element offset: 0 (lanes 0-15) / 8 (16-31)

  if (active) {
    // ---------------- GEMM1: [16x256] @ [256x128] ----------------
    #pragma unroll
    for (int nt = 0; nt < 8; ++nt)
      acc[nt] = (v8f){0.f,0.f,0.f,0.f,0.f,0.f,0.f,0.f};
    #pragma unroll
    for (int kc = 0; kc < 8; ++kc) {
      Frag a;
      const unsigned char* rp = featB + m * FEAT_ROWB + kc * 64 + off * 2;
      a.q[0] = *(const uint4*)(rp);         // K = kc*32 + off     .. +7
      a.q[1] = *(const uint4*)(rp + 32);    // K = kc*32 + 16 + off .. +7
      #pragma unroll
      for (int nt = 0; nt < 8; ++nt) {
        Frag b;
        const uint4* wp = (const uint4*)(W1sw + ((size_t)(kc * 8 + nt) * 512 + lane * 16));
        b.q[0] = wp[0]; b.q[1] = wp[1];
        acc[nt] = __builtin_amdgcn_wmma_f32_16x16x32_bf16(
            false, a.v, false, b.v, (short)0, acc[nt], false, false);
      }
    }
    // bias + relu -> bf16 h1 in LDS (C layout: lane n = L%16, rows r + 8*(L>=16))
    #pragma unroll
    for (int nt = 0; nt < 8; ++nt) {
      const int n = nt * 16 + (lane & 15);
      const float bv = b1[n];
      #pragma unroll
      for (int r = 0; r < 8; ++r) {
        const int mm = r + ((lane >> 4) << 3);
        float v = acc[nt][r] + bv;
        v = v > 0.f ? v : 0.f;
        *(unsigned short*)(h1B + mm * H1_ROWB + n * 2) = f2bf(v);
      }
    }
  }
  __syncthreads();

  if (active) {
    // ---------------- GEMM2: [16x128] @ [128x128] ----------------
    #pragma unroll
    for (int nt = 0; nt < 8; ++nt)
      acc[nt] = (v8f){0.f,0.f,0.f,0.f,0.f,0.f,0.f,0.f};
    #pragma unroll
    for (int kc = 0; kc < 4; ++kc) {
      Frag a;
      const unsigned char* rp = h1B + m * H1_ROWB + kc * 64 + off * 2;
      a.q[0] = *(const uint4*)(rp);
      a.q[1] = *(const uint4*)(rp + 32);
      #pragma unroll
      for (int nt = 0; nt < 8; ++nt) {
        Frag b;
        const uint4* wp = (const uint4*)(W2sw + ((size_t)(kc * 8 + nt) * 512 + lane * 16));
        b.q[0] = wp[0]; b.q[1] = wp[1];
        acc[nt] = __builtin_amdgcn_wmma_f32_16x16x32_bf16(
            false, a.v, false, b.v, (short)0, acc[nt], false, false);
      }
    }
    // bias + scatter-max into node buffer
    int  dmy[8];
    bool dok[8];
    #pragma unroll
    for (int r = 0; r < 8; ++r) {
      const int ge = e0 + r + ((lane >> 4) << 3);
      dok[r] = (ge < nE);
      dmy[r] = dok[r] ? dstp[ge] : 0;
    }
    #pragma unroll
    for (int nt = 0; nt < 8; ++nt) {
      const int n = nt * 16 + (lane & 15);
      const float bv = b2[n];
      #pragma unroll
      for (int r = 0; r < 8; ++r) {
        if (dok[r]) atomicMaxF(out + (size_t)dmy[r] * 128 + n, acc[nt][r] + bv);
      }
    }
  }
}

// ---------------------------------------------------------------------------
// Final layer (OUT=2): tiny GEMM -> scalar, one thread per edge, fused atomics.
// ---------------------------------------------------------------------------
__global__ void edgeconv_out_kernel(
    const float* __restrict__ h,
    const int* __restrict__ srcp, const int* __restrict__ dstp,
    const float* __restrict__ w1, const float* __restrict__ b1,
    const float* __restrict__ w2, const float* __restrict__ b2,
    float* __restrict__ out, int nE)
{
  int e = blockIdx.x * blockDim.x + threadIdx.x;
  if (e >= nE) return;
  const int i = dstp[e];
  const int j = srcp[e];
  const float* hi = h + (size_t)i * 128;
  const float* hj = h + (size_t)j * 128;
  float a0 = b1[0], a1 = b1[1];
  #pragma unroll 8
  for (int k = 0; k < 128; ++k) {
    const float xi = hi[k];
    const float d  = hj[k] - xi;
    a0 += xi * w1[k * 2]     + d * w1[(128 + k) * 2];
    a1 += xi * w1[k * 2 + 1] + d * w1[(128 + k) * 2 + 1];
  }
  a0 = fmaxf(a0, 0.f);
  a1 = fmaxf(a1, 0.f);
  const float o0 = a0 * w2[0] + a1 * w2[2] + b2[0];
  const float o1 = a0 * w2[1] + a1 * w2[3] + b2[1];
  atomicMaxF(out + (size_t)i * 2,     o0);
  atomicMaxF(out + (size_t)i * 2 + 1, o1);
}

// ---------------------------------------------------------------------------
extern "C" void kernel_launch(void* const* d_in, const int* in_sizes, int n_in,
                              void* d_out, int out_size, void* d_ws, size_t ws_size,
                              hipStream_t stream)
{
  const float* x    = (const float*)d_in[0];
  const int*   ei   = (const int*)d_in[1];     // [2, E] int32
  const float* w1_0 = (const float*)d_in[2];
  const float* b1_0 = (const float*)d_in[3];
  const float* w2_0 = (const float*)d_in[4];
  const float* b2_0 = (const float*)d_in[5];
  const float* w1_1 = (const float*)d_in[6];
  const float* b1_1 = (const float*)d_in[7];
  const float* w2_1 = (const float*)d_in[8];
  const float* b2_1 = (const float*)d_in[9];
  const float* w1_2 = (const float*)d_in[10];
  const float* b1_2 = (const float*)d_in[11];
  const float* w2_2 = (const float*)d_in[12];
  const float* b2_2 = (const float*)d_in[13];

  const int N = in_sizes[0] / 128;
  const int E = in_sizes[1] / 2;
  const int* srcp = ei;          // edge_index[0]
  const int* dstp = ei + E;      // edge_index[1]

  // Workspace layout: hA, hB node buffers (fp32) + swizzled bf16 weights.
  float* hA = (float*)d_ws;
  float* hB = hA + (size_t)N * 128;
  unsigned short* w1s0 = (unsigned short*)(hB + (size_t)N * 128);
  unsigned short* w2s0 = w1s0 + 256 * 128;
  unsigned short* w1s1 = w2s0 + 128 * 128;
  unsigned short* w2s1 = w1s1 + 256 * 128;

  // Pre-swizzle weights into WMMA fragment order (bf16).
  swizzleW_bf16<<<(256 * 128 + 255) / 256, 256, 0, stream>>>(w1_0, w1s0, 256);
  swizzleW_bf16<<<(128 * 128 + 255) / 256, 256, 0, stream>>>(w2_0, w2s0, 128);
  swizzleW_bf16<<<(256 * 128 + 255) / 256, 256, 0, stream>>>(w1_1, w1s1, 256);
  swizzleW_bf16<<<(128 * 128 + 255) / 256, 256, 0, stream>>>(w2_1, w2s1, 128);

  // Zero node buffers (0-init + atomic-max == fused relu(where(neginf,0,agg))).
  const size_t nodeElems = (size_t)N * 128;
  fill_u32<<<(unsigned)((2 * nodeElems + 255) / 256), 256, 0, stream>>>(
      (unsigned int*)hA, 0u, 2 * nodeElems);
  // d_out gets -inf init (last layer has no relu), fixed up after atomics.
  fill_u32<<<(unsigned)(((size_t)N * 2 + 255) / 256), 256, 0, stream>>>(
      (unsigned int*)d_out, 0xFF800000u, (size_t)N * 2);

  const int tiles = (E + 15) / 16;
  const int grid  = (tiles + WAVES_PER_BLOCK - 1) / WAVES_PER_BLOCK;

  edgeconv_layer_kernel<<<grid, 128, 0, stream>>>(x,  srcp, dstp, w1s0, b1_0, w2s0, b2_0, hA, E);
  edgeconv_layer_kernel<<<grid, 128, 0, stream>>>(hA, srcp, dstp, w1s1, b1_1, w2s1, b2_1, hB, E);
  edgeconv_out_kernel<<<(E + 255) / 256, 256, 0, stream>>>(hB, srcp, dstp,
      w1_2, b1_2, w2_2, b2_2, (float*)d_out, E);
  fixup_neginf<<<(N * 2 + 255) / 256, 256, 0, stream>>>((float*)d_out, N * 2);
}
